// TreeLSTM_38543036514884
// MI455X (gfx1250) — compile-verified, matrix operand order checked by reading, not run on hardware
//
// TreeLSTM + transformer child-aggregation + MTL decoders for MI455X (gfx1250).
// All GEMMs run through v_wmma_f32_16x16x32_bf16 (bf16 inputs staged in LDS,
// f32 accumulate). wave32 / WGP assumptions throughout.
// GEMM: 128x128 block tile, 8 waves in 4x2 grid, 32x64 wave tile (acc[2][4]).
#include <hip/hip_runtime.h>
#include <hip/hip_bf16.h>

// ---------------- problem constants ----------------
#define BTREES 8
#define NPT_   1093            // (3^7-1)/2
#define NTOT   (BTREES*NPT_)   // 8744
#define EDIM   256
#define HDIM   512
#define H3     1536
#define NPOIS  5000
#define NCATS  300
#define NCOOS  50

typedef __bf16 bf16_t;
typedef bf16_t bf16x8  __attribute__((ext_vector_type(8)));
typedef bf16_t bf16x16 __attribute__((ext_vector_type(16)));
typedef float  f32x8   __attribute__((ext_vector_type(8)));

__device__ __forceinline__ float sigm(float x) { return 1.0f / (1.0f + __expf(-x)); }

// ---------------- WMMA GEMM: C[M,Nout] = epi(A[M,K] @ W[Nout,K]^T + bias) ----------------
#define BM 128
#define BN 128
#define BK 32
#define LSTR 40   // bf16 elems per LDS row (80 bytes, 16B-aligned rows)

enum { EPI_NONE = 0, EPI_RELU = 1, EPI_RES = 2, EPI_POI = 3, EPI_CATP = 4, EPI_COOP = 5 };

__device__ __forceinline__ bf16x8 cvt8(float4 f0, float4 f1)
{
    bf16x8 p;
    p[0] = (bf16_t)f0.x; p[1] = (bf16_t)f0.y; p[2] = (bf16_t)f0.z; p[3] = (bf16_t)f0.w;
    p[4] = (bf16_t)f1.x; p[5] = (bf16_t)f1.y; p[6] = (bf16_t)f1.z; p[7] = (bf16_t)f1.w;
    return p;
}

__global__ __launch_bounds__(256) void treelstm_gemm_wmma(
    const float* __restrict__ A, const float* __restrict__ W,
    const float* __restrict__ bias, const float* __restrict__ resid,
    const float* __restrict__ mtl, float* __restrict__ C,
    int M, int Nout, int K, int mode)
{
    __shared__ bf16_t sA[BM * LSTR];
    __shared__ bf16_t sB[BN * LSTR];

    const int tid  = threadIdx.x;
    const int bR   = blockIdx.y, bCo = blockIdx.x;
    const int row0 = bR * BM, col0 = bCo * BN;

    const int wid  = tid >> 5, lane = tid & 31;
    const int wr   = wid >> 1, wc = wid & 1;        // 4x2 wave grid; wave tile 32x64
    const int half = lane >> 4, l16 = lane & 15;
    const int hb   = half * 8;

    const f32x8 vzero = {0.f,0.f,0.f,0.f,0.f,0.f,0.f,0.f};
    f32x8 acc[2][4];
    #pragma unroll
    for (int i = 0; i < 2; ++i)
        #pragma unroll
        for (int j = 0; j < 4; ++j) acc[i][j] = vzero;

    // staging maps: 128 rows x 32 cols, 16 floats per thread for both tiles
    const int srow = tid >> 1, sc0 = (tid & 1) * 16;

    // block-uniform predicates (scalar branches only)
    const bool vec4  = ((K & 3) == 0);
    const bool fullA = (row0 + BM <= M) && vec4;
    const bool fullB = (col0 + BN <= Nout) && vec4;
    const int  garc  = (row0 + srow < M) ? (row0 + srow) : (M - 1);
    const int  gbrc  = (col0 + srow < Nout) ? (col0 + srow) : (Nout - 1);

    for (int kt = 0; kt < K; kt += BK) {
        const bool fullK = (kt + BK <= K);
        // ---- stage A tile (f32 -> bf16, vector path for interior blocks) ----
        {
            const float* ap = A + (size_t)garc * K + kt + sc0;
            bf16x8 plo, phi;
            if (fullA && fullK) {
                const float4* p4 = (const float4*)ap;
                plo = cvt8(p4[0], p4[1]);
                phi = cvt8(p4[2], p4[3]);
                if (kt + BK < K) __builtin_prefetch(ap + BK, 0, 1);
            } else {
                int gr = row0 + srow;
                #pragma unroll
                for (int e = 0; e < 8; ++e) {
                    int col  = kt + sc0 + e;
                    int colc = (col < K) ? col : (K - 1);
                    float v  = A[(size_t)garc * K + colc];
                    plo[e]   = (bf16_t)((gr < M && col < K) ? v : 0.f);
                }
                #pragma unroll
                for (int e = 0; e < 8; ++e) {
                    int col  = kt + sc0 + 8 + e;
                    int colc = (col < K) ? col : (K - 1);
                    float v  = A[(size_t)garc * K + colc];
                    phi[e]   = (bf16_t)((gr < M && col < K) ? v : 0.f);
                }
            }
            *(bf16x8*)&sA[srow * LSTR + sc0]     = plo;
            *(bf16x8*)&sA[srow * LSTR + sc0 + 8] = phi;
        }
        // ---- stage W tile ----
        {
            const float* wp = W + (size_t)gbrc * K + kt + sc0;
            bf16x8 plo, phi;
            if (fullB && fullK) {
                const float4* p4 = (const float4*)wp;
                plo = cvt8(p4[0], p4[1]);
                phi = cvt8(p4[2], p4[3]);
                if (kt + BK < K) __builtin_prefetch(wp + BK, 0, 1);
            } else {
                int gr = col0 + srow;
                #pragma unroll
                for (int e = 0; e < 8; ++e) {
                    int col  = kt + sc0 + e;
                    int colc = (col < K) ? col : (K - 1);
                    float v  = W[(size_t)gbrc * K + colc];
                    plo[e]   = (bf16_t)((gr < Nout && col < K) ? v : 0.f);
                }
                #pragma unroll
                for (int e = 0; e < 8; ++e) {
                    int col  = kt + sc0 + 8 + e;
                    int colc = (col < K) ? col : (K - 1);
                    float v  = W[(size_t)gbrc * K + colc];
                    phi[e]   = (bf16_t)((gr < Nout && col < K) ? v : 0.f);
                }
            }
            *(bf16x8*)&sB[srow * LSTR + sc0]     = plo;
            *(bf16x8*)&sB[srow * LSTR + sc0 + 8] = phi;
        }
        __syncthreads();

        // ---- fragment gathers per documented 16-bit A layout ----
        bf16x16 afr[2], bfr[4];
        #pragma unroll
        for (int i = 0; i < 2; ++i) {
            int r = wr * 32 + i * 16 + l16;
            bf16x8 a0 = *(const bf16x8*)&sA[r * LSTR + hb];
            bf16x8 a1 = *(const bf16x8*)&sA[r * LSTR + 16 + hb];
            afr[i] = __builtin_shufflevector(a0, a1, 0,1,2,3,4,5,6,7,8,9,10,11,12,13,14,15);
        }
        #pragma unroll
        for (int j = 0; j < 4; ++j) {
            int r = wc * 64 + j * 16 + l16;
            bf16x8 b0 = *(const bf16x8*)&sB[r * LSTR + hb];
            bf16x8 b1 = *(const bf16x8*)&sB[r * LSTR + 16 + hb];
            bfr[j] = __builtin_shufflevector(b0, b1, 0,1,2,3,4,5,6,7,8,9,10,11,12,13,14,15);
        }
        #pragma unroll
        for (int i = 0; i < 2; ++i)
            #pragma unroll
            for (int j = 0; j < 4; ++j)
                acc[i][j] = __builtin_amdgcn_wmma_f32_16x16x32_bf16(
                    false, afr[i], false, bfr[j], (short)0, acc[i][j], false, false);
        __syncthreads();
    }

    // ---- epilogue ----
    float m0 = 0.f, m1 = 0.f, m2 = 0.f;
    if (mtl) { m0 = mtl[0]; m1 = mtl[1]; m2 = mtl[2]; }
    const float a0 = 0.5f * __expf(-m0), a1 = 0.5f * __expf(-m1), a2 = 0.5f * __expf(-m2);
    const float cconst = m0 + m1 + m2;
    const bool fullC = (row0 + BM <= M) && (col0 + BN <= Nout);

    #pragma unroll
    for (int i = 0; i < 2; ++i) {
        #pragma unroll
        for (int j = 0; j < 4; ++j) {
            int col = col0 + wc * 64 + j * 16 + l16;
            if (!fullC && col >= Nout) continue;
            float bv = bias ? bias[col] : 0.f;
            #pragma unroll
            for (int r = 0; r < 8; ++r) {
                int row = row0 + wr * 32 + i * 16 + half * 8 + r;
                if (!fullC && row >= M) continue;
                size_t idx = (size_t)row * Nout + col;
                float v = acc[i][j][r] + bv;
                if      (mode == EPI_RELU) v = fmaxf(v, 0.f);
                else if (mode == EPI_RES)  v += resid[idx];
                else if (mode == EPI_POI)  v = a0 * v + cconst;
                else if (mode == EPI_CATP) v = C[idx] + a1 * v;
                else if (mode == EPI_COOP) v = C[idx] + a2 * v;
                C[idx] = v;
            }
        }
    }
}

// ---------------- embedding gather + mask ----------------
__global__ __launch_bounds__(256) void emb_gather_kernel(
    const int* __restrict__ uids, const int* __restrict__ fids,
    const int* __restrict__ tids, const int* __restrict__ mask,
    const float* __restrict__ user_emb, const float* __restrict__ fuse_emb,
    const float* __restrict__ time_pe, float* __restrict__ emb)
{
    int row = blockIdx.x, d = threadIdx.x;
    int m = mask[row];
    int uid = uids[row] * m, fid = fids[row] * m, tid = tids[row] * m;
    float e = (d < 128) ? user_emb[(size_t)uid * 128 + d]
                        : fuse_emb[(size_t)fid * 128 + (d - 128)];
    e += 0.5f * time_pe[(size_t)tid * EDIM + d];
    emb[(size_t)row * EDIM + d] = e * (float)m;
}

// ---------------- 3-token MHA (per sequence, per head) ----------------
__global__ __launch_bounds__(256) void attn3_kernel(
    const float* __restrict__ qkv, float* __restrict__ out)
{
    int m = blockIdx.x >> 1, head = blockIdx.x & 1;
    int d = threadIdx.x;                 // 0..255 = head dim
    const float* base = qkv + (size_t)(3 * m) * H3 + head * 256;
    float q[3], k[3], v[3];
    #pragma unroll
    for (int i = 0; i < 3; ++i) {
        q[i] = base[(size_t)i * H3 + d] * 0.0625f;   // 1/sqrt(256)
        k[i] = base[(size_t)i * H3 + 512 + d];
        v[i] = base[(size_t)i * H3 + 1024 + d];
    }
    float p[9];
    #pragma unroll
    for (int i = 0; i < 3; ++i)
        #pragma unroll
        for (int j = 0; j < 3; ++j) p[3 * i + j] = q[i] * k[j];
    #pragma unroll
    for (int off = 16; off > 0; off >>= 1)
        #pragma unroll
        for (int t = 0; t < 9; ++t) p[t] += __shfl_down(p[t], off, 32);

    __shared__ float red[8][9];
    __shared__ float aw[9];
    int lane = threadIdx.x & 31, wid = threadIdx.x >> 5;
    if (lane == 0)
        #pragma unroll
        for (int t = 0; t < 9; ++t) red[wid][t] = p[t];
    __syncthreads();
    if (threadIdx.x < 3) {
        int i = threadIdx.x;
        float s0 = 0.f, s1 = 0.f, s2 = 0.f;
        for (int w = 0; w < 8; ++w) { s0 += red[w][3*i]; s1 += red[w][3*i+1]; s2 += red[w][3*i+2]; }
        float mx = fmaxf(s0, fmaxf(s1, s2));
        float e0 = __expf(s0 - mx), e1 = __expf(s1 - mx), e2 = __expf(s2 - mx);
        float inv = 1.f / (e0 + e1 + e2);
        aw[3*i] = e0 * inv; aw[3*i+1] = e1 * inv; aw[3*i+2] = e2 * inv;
    }
    __syncthreads();
    #pragma unroll
    for (int i = 0; i < 3; ++i)
        out[(size_t)(3 * m + i) * HDIM + head * 256 + d] =
            aw[3*i] * v[0] + aw[3*i+1] * v[1] + aw[3*i+2] * v[2];
}

// ---------------- LayerNorm (row=512), out = LN(x + res) ----------------
__global__ __launch_bounds__(256) void ln_kernel(
    const float* __restrict__ x, const float* __restrict__ res,
    const float* __restrict__ w, const float* __restrict__ b,
    float* __restrict__ out)
{
    int row = blockIdx.x, t = threadIdx.x;
    const float* xr = x + (size_t)row * HDIM;
    const float* rr = res + (size_t)row * HDIM;
    float v0 = xr[t] + rr[t];
    float v1 = xr[t + 256] + rr[t + 256];

    __shared__ float sred[8];
    __shared__ float sstat;
    int lane = t & 31, wid = t >> 5;

    float s = v0 + v1;
    #pragma unroll
    for (int off = 16; off > 0; off >>= 1) s += __shfl_down(s, off, 32);
    if (!lane) sred[wid] = s;
    __syncthreads();
    if (t == 0) { float tot = 0.f; for (int i = 0; i < 8; ++i) tot += sred[i]; sstat = tot * (1.f/512.f); }
    __syncthreads();
    float mean = sstat;
    float d0 = v0 - mean, d1 = v1 - mean;
    float s2 = d0 * d0 + d1 * d1;
    #pragma unroll
    for (int off = 16; off > 0; off >>= 1) s2 += __shfl_down(s2, off, 32);
    __syncthreads();
    if (!lane) sred[wid] = s2;
    __syncthreads();
    if (t == 0) { float tot = 0.f; for (int i = 0; i < 8; ++i) tot += sred[i]; sstat = tot * (1.f/512.f); }
    __syncthreads();
    float inv = rsqrtf(sstat + 1e-5f);
    out[(size_t)row * HDIM + t]       = d0 * inv * w[t] + b[t];
    out[(size_t)row * HDIM + t + 256] = d1 * inv * w[t + 256] + b[t + 256];
}

// ---------------- gather children h -> z tokens ----------------
__global__ __launch_bounds__(256) void copy_hch_kernel(
    const float* __restrict__ h, float* __restrict__ z, int n, int cs)
{
    int r = blockIdx.x;
    int t = r / (3 * n), p = r % (3 * n);
    size_t src = ((size_t)t * NPT_ + cs + p) * HDIM;
    size_t dst = (size_t)r * HDIM;
    z[dst + threadIdx.x]       = h[src + threadIdx.x];
    z[dst + threadIdx.x + 256] = h[src + threadIdx.x + 256];
}

// ---------------- gather xl = x[level slice] ----------------
__global__ __launch_bounds__(256) void gather_xl_kernel(
    const float* __restrict__ emb, float* __restrict__ xl, int n, int st)
{
    int r = blockIdx.x, t = r / n, j = r % n;
    xl[(size_t)r * EDIM + threadIdx.x] =
        emb[((size_t)t * NPT_ + st + j) * EDIM + threadIdx.x];
}

// ---------------- leaf LSTM cell ----------------
__global__ __launch_bounds__(256) void leaf_gate_kernel(
    const float* __restrict__ iou0, const float* __restrict__ c0,
    float* __restrict__ h, float* __restrict__ c)
{
    int node = blockIdx.x;
    int t = node / 729, j = node % 729;
    size_t g = (size_t)t * NPT_ + 364 + j;
    for (int d = threadIdx.x; d < HDIM; d += 256) {
        float iv = iou0[g * H3 + d];
        float ov = iou0[g * H3 + 512 + d];
        float uv = iou0[g * H3 + 1024 + d];
        float cn = sigm(iv) * tanhf(uv) + c0[g * HDIM + d];
        float hn = sigm(ov) * tanhf(cn);
        c[g * HDIM + d] = cn;
        h[g * HDIM + d] = hn;
    }
}

// ---------------- interior LSTM cell (forget-gated child-sum) ----------------
__global__ __launch_bounds__(256) void interior_gate_kernel(
    const float* __restrict__ iou, const float* __restrict__ wx,
    const float* __restrict__ hc, float* __restrict__ h, float* __restrict__ c,
    int n, int st, int cs)
{
    int r = blockIdx.x;
    int t = r / n, j = r % n;
    size_t gp = (size_t)t * NPT_ + st + j;
    size_t gc = (size_t)t * NPT_ + cs + 3 * (size_t)j;
    for (int d = threadIdx.x; d < HDIM; d += 256) {
        float iv = iou[(size_t)r * H3 + d];
        float ov = iou[(size_t)r * H3 + 512 + d];
        float uv = iou[(size_t)r * H3 + 1024 + d];
        float wxv = wx[(size_t)r * HDIM + d];
        float cpre = 0.f;
        #pragma unroll
        for (int k = 0; k < 3; ++k) {
            float f = sigm(wxv + hc[(size_t)r * H3 + k * 512 + d]);
            cpre += f * c[(gc + k) * HDIM + d];
        }
        float cn = sigm(iv) * tanhf(uv) + cpre;
        float hn = sigm(ov) * tanhf(cn);
        c[gp * HDIM + d] = cn;
        h[gp * HDIM + d] = hn;
    }
}

// ---------------- host driver ----------------
extern "C" void kernel_launch(void* const* d_in, const int* in_sizes, int n_in,
                              void* d_out, int out_size, void* d_ws, size_t ws_size,
                              hipStream_t stream)
{
    (void)in_sizes; (void)n_in; (void)out_size; (void)ws_size;
    const int*   user_ids = (const int*)d_in[0];
    const int*   feat_ids = (const int*)d_in[1];
    const int*   time_ids = (const int*)d_in[2];
    const int*   maskp    = (const int*)d_in[3];
    const float* user_emb = (const float*)d_in[6];
    const float* fuse_emb = (const float*)d_in[7];
    const float* time_pe  = (const float*)d_in[8];
    const float* Wf   = (const float*)d_in[9];
    const float* bfv  = (const float*)d_in[10];
    const float* Wiou = (const float*)d_in[11];
    const float* Uiou = (const float*)d_in[12];
    const float* biou = (const float*)d_in[13];
    const float* aiw  = (const float*)d_in[14];
    const float* aib  = (const float*)d_in[15];
    const float* aow  = (const float*)d_in[16];
    const float* aob  = (const float*)d_in[17];
    const float* f1w  = (const float*)d_in[18];
    const float* f1b  = (const float*)d_in[19];
    const float* f2w  = (const float*)d_in[20];
    const float* f2b  = (const float*)d_in[21];
    const float* l1w  = (const float*)d_in[22];
    const float* l1b  = (const float*)d_in[23];
    const float* l2w  = (const float*)d_in[24];
    const float* l2b  = (const float*)d_in[25];
    const float* dpw  = (const float*)d_in[26];
    const float* dpb  = (const float*)d_in[27];
    const float* dcw  = (const float*)d_in[28];
    const float* dcb  = (const float*)d_in[29];
    const float* dow  = (const float*)d_in[30];
    const float* dob  = (const float*)d_in[31];
    const float* c2pw = (const float*)d_in[32];
    const float* c2pb = (const float*)d_in[33];
    const float* o2pw = (const float*)d_in[34];
    const float* o2pb = (const float*)d_in[35];
    const float* mtl  = (const float*)d_in[36];
    const float* c0   = (const float*)d_in[37];

    float* out = (float*)d_out;
    float* ws  = (float*)d_ws;

    // workspace carve (floats); reused across sides
    size_t off = 0;
    float* emb  = ws + off; off += (size_t)NTOT * EDIM;
    float* h    = ws + off; off += (size_t)NTOT * HDIM;
    float* c    = ws + off; off += (size_t)NTOT * HDIM;
    float* iou0 = ws + off; off += (size_t)NTOT * H3;   // reused as qkv after leaves
    float* z    = ws + off; off += (size_t)5832 * HDIM;
    float* z1   = ws + off; off += (size_t)5832 * HDIM;
    float* ao   = ws + off; off += (size_t)5832 * HDIM;
    float* ta   = ws + off; off += (size_t)5832 * HDIM;
    float* xl   = ws + off; off += (size_t)1944 * EDIM;
    float* wx   = ws + off; off += (size_t)1944 * HDIM;
    float* ioub = ws + off; off += (size_t)1944 * H3;
    float* qkv  = iou0;

    auto gemm = [&](const float* A, const float* W, const float* bias,
                    const float* resid, const float* mtlp, float* C_,
                    int M, int Nout, int K, int mode) {
        dim3 grid((Nout + BN - 1) / BN, (M + BM - 1) / BM);
        treelstm_gemm_wmma<<<grid, 256, 0, stream>>>(A, W, bias, resid, mtlp, C_,
                                                     M, Nout, K, mode);
    };

    const int pow3[7] = {1, 3, 9, 27, 81, 243, 729};
    const int offs[8] = {0, 1, 4, 13, 40, 121, 364, 1093};

    for (int s = 0; s < 2; ++s) {
        const int* uid_s = user_ids + (size_t)s * NTOT;
        const int* fid_s = feat_ids + (size_t)s * NTOT;
        const int* tid_s = time_ids + (size_t)s * NTOT;
        const int* msk_s = maskp    + (size_t)s * NTOT;
        const float* ue_s = user_emb + (size_t)s * 2000 * 128;
        const float* fe_s = fuse_emb + (size_t)s * 5350 * 128;
        const float* tp_s = time_pe  + (size_t)s * 600 * EDIM;
        const float* Wf_s   = Wf   + (size_t)s * HDIM * EDIM;
        const float* bf_s   = bfv  + (size_t)s * HDIM;
        const float* Wiou_s = Wiou + (size_t)s * H3 * EDIM;
        const float* Uiou_s = Uiou + (size_t)s * H3 * H3;
        const float* biou_s = biou + (size_t)s * H3;
        const float* mtl_s  = mtl  + (size_t)s * 3;
        const float* c0_s   = c0   + (size_t)s * NTOT * HDIM;

        // 1) embedding gather (mask folded in)
        emb_gather_kernel<<<NTOT, 256, 0, stream>>>(uid_s, fid_s, tid_s, msk_s,
                                                    ue_s, fe_s, tp_s, emb);
        // 2) iou0 = emb @ Wiou^T   (mask already in emb; no bias at leaves)
        gemm(emb, Wiou_s, nullptr, nullptr, nullptr, iou0, NTOT, H3, EDIM, EPI_NONE);
        // 3) leaf cells (level 6)
        leaf_gate_kernel<<<BTREES * 729, 256, 0, stream>>>(iou0, c0_s, h, c);

        // 4) bottom-up interior levels
        for (int l = 5; l >= 0; --l) {
            int n = pow3[l], st = offs[l], cs = offs[l + 1];
            int Mn = BTREES * n, R = 3 * Mn;
            copy_hch_kernel<<<R, 256, 0, stream>>>(h, z, n, cs);
            for (int L = 0; L < 2; ++L) {
                size_t li = (size_t)(s * 2 + L);
                const float* aiw_l = aiw + li * H3 * HDIM;
                const float* aib_l = aib + li * H3;
                const float* aow_l = aow + li * HDIM * HDIM;
                const float* aob_l = aob + li * HDIM;
                const float* f1w_l = f1w + li * HDIM * HDIM;
                const float* f1b_l = f1b + li * HDIM;
                const float* f2w_l = f2w + li * HDIM * HDIM;
                const float* f2b_l = f2b + li * HDIM;
                const float* l1w_l = l1w + li * HDIM;
                const float* l1b_l = l1b + li * HDIM;
                const float* l2w_l = l2w + li * HDIM;
                const float* l2b_l = l2b + li * HDIM;

                gemm(z, aiw_l, aib_l, nullptr, nullptr, qkv, R, H3, HDIM, EPI_NONE);
                attn3_kernel<<<Mn * 2, 256, 0, stream>>>(qkv, ao);
                gemm(ao, aow_l, aob_l, nullptr, nullptr, ta, R, HDIM, HDIM, EPI_NONE);
                ln_kernel<<<R, 256, 0, stream>>>(z, ta, l1w_l, l1b_l, z1);
                gemm(z1, f1w_l, f1b_l, nullptr, nullptr, ao, R, HDIM, HDIM, EPI_RELU);
                gemm(ao, f2w_l, f2b_l, nullptr, nullptr, ta, R, HDIM, HDIM, EPI_NONE);
                ln_kernel<<<R, 256, 0, stream>>>(z1, ta, l2w_l, l2b_l, z);
            }
            // hc == z viewed as [Mn, 1536] (contiguous)
            gather_xl_kernel<<<Mn, 256, 0, stream>>>(emb, xl, n, st);
            gemm(z, Uiou_s, biou_s, nullptr, nullptr, ioub, Mn, H3, H3, EPI_NONE);
            gemm(xl, Wiou_s, nullptr, ioub, nullptr, ioub, Mn, H3, EDIM, EPI_RES);
            gemm(xl, Wf_s, bf_s, nullptr, nullptr, wx, Mn, HDIM, EDIM, EPI_NONE);
            interior_gate_kernel<<<Mn, 256, 0, stream>>>(ioub, wx, z, h, c, n, st, cs);
        }

        // 5) decoders + multi-task fusion into d_out
        float* out_poi = out + (size_t)s * NTOT * (NPOIS + NCATS + NCOOS);
        float* out_cat = out_poi + (size_t)NTOT * NPOIS;
        float* out_coo = out_cat + (size_t)NTOT * NCATS;
        const float* dpw_s = dpw + (size_t)s * NPOIS * HDIM;
        const float* dpb_s = dpb + (size_t)s * NPOIS;
        const float* dcw_s = dcw + (size_t)s * NCATS * HDIM;
        const float* dcb_s = dcb + (size_t)s * NCATS;
        const float* dow_s = dow + (size_t)s * NCOOS * HDIM;
        const float* dob_s = dob + (size_t)s * NCOOS;
        const float* c2pw_s = c2pw + (size_t)s * NPOIS * NCATS;
        const float* c2pb_s = c2pb + (size_t)s * NPOIS;
        const float* o2pw_s = o2pw + (size_t)s * NPOIS * NCOOS;
        const float* o2pb_s = o2pb + (size_t)s * NPOIS;

        gemm(h, dcw_s, dcb_s, nullptr, nullptr, out_cat, NTOT, NCATS, HDIM, EPI_NONE);
        gemm(h, dow_s, dob_s, nullptr, nullptr, out_coo, NTOT, NCOOS, HDIM, EPI_NONE);
        gemm(h, dpw_s, dpb_s, nullptr, mtl_s, out_poi, NTOT, NPOIS, HDIM, EPI_POI);
        gemm(out_cat, c2pw_s, c2pb_s, nullptr, mtl_s, out_poi, NTOT, NPOIS, NCATS, EPI_CATP);
        gemm(out_coo, o2pw_s, o2pb_s, nullptr, mtl_s, out_poi, NTOT, NPOIS, NCOOS, EPI_COOP);
    }
}
